// minGRUDepthController_1391569403954
// MI455X (gfx1250) — matrix operand order repaired and checked by reading, not compiled, not run on hardware
//
#include <hip/hip_runtime.h>
#include <math.h>

#define D_MODEL 1024
#define D_CTRL  1024
#define D_CAT   2048
#define N_HEADS 16
#define TOPK    4
#define BB      8
#define SS      2048
#define LN_EPS  1e-5f
#define ALPHA_N (BB*SS*N_HEADS)      /* 262144 */
#define HNEW_N  (BB*D_CTRL)          /* 8192   */
#define ENT_OFF (ALPHA_N + HNEW_N)   /* 270336 */

#define BROW    2064                 /* padded LDS row stride (bytes) for B tiles */

typedef __attribute__((ext_vector_type(16))) __bf16 v16bf;
typedef __attribute__((ext_vector_type(8)))  float  v8f;

union FragU { uint4 q[2]; v16bf v; };

__device__ __forceinline__ unsigned short f2bf(float f) {
  unsigned u = __float_as_uint(f);
  u += 0x7FFFu + ((u >> 16) & 1u);          // round-to-nearest-even
  return (unsigned short)(u >> 16);
}

// ---------------------------------------------------------------- init
__global__ void init_kernel(float* __restrict__ g_sum, float* __restrict__ out) {
  int i = blockIdx.x * 256 + threadIdx.x;
  if (i < BB * D_MODEL) g_sum[i] = 0.0f;
  if (i == 0) out[ENT_OFF] = 0.0f;
}

// ------------------------------------------------- r_w1 fp32 -> bf16
__global__ void w1cvt_kernel(const float* __restrict__ w, unsigned short* __restrict__ wb) {
  int i = (blockIdx.x * 256 + threadIdx.x) * 8;
  #pragma unroll
  for (int j = 0; j < 8; ++j) wb[i + j] = f2bf(w[i + j]);
}

// --------------------------------- r_w2 fp32 [N,H] -> bf16 transposed [H,N]
__global__ void w2cvt_kernel(const float* __restrict__ w, unsigned short* __restrict__ wt) {
  int i = blockIdx.x * 256 + threadIdx.x;        // i = h*1024 + n
  int h = i >> 10, n = i & 1023;
  wt[i] = f2bf(w[n * N_HEADS + h]);
}

// ---------------------------------------------------------------- pool
__global__ void pool_kernel(const float* __restrict__ x,
                            const float* __restrict__ pg,
                            const float* __restrict__ pb,
                            float* __restrict__ g_sum) {
  __shared__ float acc[D_MODEL];
  const int b   = blockIdx.y;
  const int s0  = blockIdx.x * 64;
  const int tid = threadIdx.x, lane = tid & 31, wv = tid >> 5;
  for (int i = tid; i < D_MODEL; i += 256) acc[i] = 0.0f;
  __syncthreads();
  for (int t = 0; t < 8; ++t) {
    const int s = s0 + wv * 8 + t;
    const float* row = x + ((size_t)(b * SS + s)) * D_MODEL;
    float xv[32], sum = 0.f, sq = 0.f;
    #pragma unroll
    for (int j = 0; j < 32; ++j) { float v = row[lane + 32 * j]; xv[j] = v; sum += v; sq += v * v; }
    #pragma unroll
    for (int o = 16; o > 0; o >>= 1) { sum += __shfl_xor(sum, o, 32); sq += __shfl_xor(sq, o, 32); }
    const float m  = sum * (1.0f / D_MODEL);
    const float rs = rsqrtf(sq * (1.0f / D_MODEL) - m * m + LN_EPS);
    #pragma unroll
    for (int j = 0; j < 32; ++j) {
      const int d = lane + 32 * j;
      atomicAdd(&acc[d], (xv[j] - m) * rs * pg[d] + pb[d]);
    }
  }
  __syncthreads();
  for (int i = tid; i < D_MODEL; i += 256) atomicAdd(&g_sum[b * D_MODEL + i], acc[i]);
}

// ---------------------------------------------------------------- ctrl
__global__ void ctrl_kernel(const float* __restrict__ g_sum,
                            const float* __restrict__ state,
                            const float* __restrict__ inp_w, const float* __restrict__ inp_b,
                            const float* __restrict__ wz_w,  const float* __restrict__ wz_b,
                            const float* __restrict__ wh_w,  const float* __restrict__ wh_b,
                            const float* __restrict__ lnh_g, const float* __restrict__ lnh_b,
                            float* __restrict__ out, float* __restrict__ h_ws) {
  __shared__ float gl[D_CTRL];
  __shared__ float xc[D_CTRL];
  __shared__ float red[256];
  const int b = blockIdx.x, tid = threadIdx.x;
  #pragma unroll
  for (int i = 0; i < 4; ++i) gl[tid + 256 * i] = g_sum[b * D_MODEL + tid + 256 * i] * (1.0f / SS);
  __syncthreads();
  float a0[4];
  #pragma unroll
  for (int i = 0; i < 4; ++i) a0[i] = inp_b[tid + 256 * i];
  for (int k = 0; k < D_MODEL; ++k) {
    const float gk = gl[k];
    const float* r = inp_w + (size_t)k * D_CTRL;
    #pragma unroll
    for (int i = 0; i < 4; ++i) a0[i] = fmaf(gk, r[tid + 256 * i], a0[i]);
  }
  #pragma unroll
  for (int i = 0; i < 4; ++i) xc[tid + 256 * i] = a0[i];
  __syncthreads();
  float az[4], ah[4];
  #pragma unroll
  for (int i = 0; i < 4; ++i) { az[i] = wz_b[tid + 256 * i]; ah[i] = wh_b[tid + 256 * i]; }
  for (int k = 0; k < D_CTRL; ++k) {
    const float xk = xc[k];
    const float* rz = wz_w + (size_t)k * D_CTRL;
    const float* rh = wh_w + (size_t)k * D_CTRL;
    #pragma unroll
    for (int i = 0; i < 4; ++i) {
      az[i] = fmaf(xk, rz[tid + 256 * i], az[i]);
      ah[i] = fmaf(xk, rh[tid + 256 * i], ah[i]);
    }
  }
  float hn[4], ps = 0.f, pq = 0.f;
  #pragma unroll
  for (int i = 0; i < 4; ++i) {
    const int j = tid + 256 * i;
    const float z  = 1.0f / (1.0f + expf(-az[i]));
    const float hc = tanhf(ah[i]);
    hn[i] = (1.0f - z) * state[b * D_CTRL + j] + z * hc;
    ps += hn[i]; pq += hn[i] * hn[i];
  }
  red[tid] = ps; __syncthreads();
  for (int s = 128; s > 0; s >>= 1) { if (tid < s) red[tid] += red[tid + s]; __syncthreads(); }
  const float m = red[0] * (1.0f / D_CTRL);
  __syncthreads();
  red[tid] = pq; __syncthreads();
  for (int s = 128; s > 0; s >>= 1) { if (tid < s) red[tid] += red[tid + s]; __syncthreads(); }
  const float rs = rsqrtf(red[0] * (1.0f / D_CTRL) - m * m + LN_EPS);
  #pragma unroll
  for (int i = 0; i < 4; ++i) {
    const int j = tid + 256 * i;
    const float v = (hn[i] - m) * rs * lnh_g[j] + lnh_b[j];
    out[ALPHA_N + b * D_CTRL + j] = v;
    h_ws[b * D_CTRL + j] = v;
  }
}

// -------------------------------------------------------------- router
__global__ void router_kernel(const float* __restrict__ x,
                              const float* __restrict__ h_ws,
                              const float* __restrict__ rln_g, const float* __restrict__ rln_b,
                              const unsigned short* __restrict__ w1bf,
                              const float* __restrict__ r_b1,
                              const unsigned short* __restrict__ w2t,
                              const float* __restrict__ r_b2,
                              float* __restrict__ out) {
  __shared__ __align__(16) unsigned char bBuf[2][32 * BROW];   // async double buffer (132 KB)
  __shared__ __align__(16) float hn_s[D_CTRL];
  __shared__ float mu_s[16], rsd_s[16];
  __shared__ __align__(16) unsigned short aTile[16 * 48];      // 16B-aligned rows
  __shared__ __align__(16) unsigned short hidbuf[16 * 1024];
  __shared__ float logits_s[16 * 16];
  const int b   = blockIdx.y;
  const int s0  = blockIdx.x * 16;
  const int tid = threadIdx.x, lane = tid & 31, wv = tid >> 5;

  // async copy of w1 K-chunk [krow..krow+32) x 1024 cols into padded LDS buffer.
  // "+v" ties the addresses into a serial register chain: nothing to hoist/spill.
  auto asyncB = [&](unsigned dstBase, int krow) {
    const int r0 = tid >> 7, cb = (tid & 127) * 16;
    const unsigned char* g = (const unsigned char*)w1bf +
        (size_t)(krow + r0) * (D_CTRL * 2) + (size_t)cb;
    unsigned d = dstBase + (unsigned)(r0 * BROW + cb);
    #pragma unroll
    for (int i = 0; i < 16; ++i) {
      asm volatile("global_load_async_to_lds_b128 %0, %1, off"
                   : "+v"(d), "+v"(g) :: "memory");
      d += 2 * BROW;
      g += 2 * (D_CTRL * 2);
    }
  };
  const unsigned b0base = (unsigned)(unsigned long long)(const void*)&bBuf[0][0];
  const unsigned b1base = (unsigned)(unsigned long long)(const void*)&bBuf[1][0];

  // h_new row (4 KB) first, then prefetch B chunk 0 behind it
  {
    unsigned hdst = (unsigned)(unsigned long long)(const void*)&hn_s[0] + (unsigned)(tid * 16);
    const void* gsrc = (const void*)(h_ws + b * D_CTRL + tid * 4);
    asm volatile("global_load_async_to_lds_b128 %0, %1, off" :: "v"(hdst), "v"(gsrc) : "memory");
  }
  asyncB(b0base, 0);
  // ASYNC loads complete in order: <=16 outstanding  =>  hn copy done
  asm volatile("s_wait_asynccnt 0x10" ::: "memory");
  __syncthreads();

  // LN stats over concat(x_tok, h): 2 tokens per wave (overlaps B chunk-0 DMA)
  for (int t = 0; t < 2; ++t) {
    const int m = wv * 2 + t;
    const float* row = x + ((size_t)(b * SS + s0 + m)) * D_MODEL;
    float sum = 0.f, sq = 0.f;
    #pragma unroll
    for (int j = 0; j < 32; ++j) { float v = row[lane + 32 * j];  sum += v; sq += v * v; }
    #pragma unroll
    for (int j = 0; j < 32; ++j) { float v = hn_s[lane + 32 * j]; sum += v; sq += v * v; }
    #pragma unroll
    for (int o = 16; o > 0; o >>= 1) { sum += __shfl_xor(sum, o, 32); sq += __shfl_xor(sq, o, 32); }
    if (lane == 0) {
      const float mm = sum * (1.0f / D_CAT);
      mu_s[m]  = mm;
      rsd_s[m] = rsqrtf(sq * (1.0f / D_CAT) - mm * mm + LN_EPS);
    }
  }
  __syncthreads();

  const int nbase = wv * 128;
  float bias[8];
  #pragma unroll
  for (int ni = 0; ni < 8; ++ni) bias[ni] = r_b1[nbase + ni * 16 + (lane & 15)];

  v8f acc[8] = {};
  const int arow = lane & 15;          // A fragment: M row
  const int koff = (lane >> 4) * 8;    // lanes 0-15: K {0-7,16-23}; 16-31: K {8-15,24-31}

  auto stageA = [&](int k0) {
    if (k0 < D_MODEL) {                       // chunk entirely in x half -> pure global loads
      #pragma unroll
      for (int i = 0; i < 2; ++i) {
        const int e = tid + 256 * i, m = e >> 5, kk = e & 31, k = k0 + kk;
        const float v = x[((size_t)(b * SS + s0 + m)) * D_MODEL + k];
        aTile[m * 48 + kk] = f2bf((v - mu_s[m]) * rsd_s[m] * rln_g[k] + rln_b[k]);
      }
    } else {                                  // chunk entirely in h half -> pure LDS loads
      #pragma unroll
      for (int i = 0; i < 2; ++i) {
        const int e = tid + 256 * i, m = e >> 5, kk = e & 31, k = k0 + kk;
        const float v = hn_s[k - D_MODEL];
        aTile[m * 48 + kk] = f2bf((v - mu_s[m]) * rsd_s[m] * rln_g[k] + rln_b[k]);
      }
    }
  };
  auto loadA = [&]() {
    FragU a;
    const unsigned short* ap = &aTile[arow * 48 + koff];
    a.q[0] = *(const uint4*)(ap);
    a.q[1] = *(const uint4*)(ap + 16);
    return a;
  };
  auto chunk = [&](const unsigned char* curPtr, unsigned nxtBase, int k0, bool last) {
    if (!last) asyncB(nxtBase, k0 + 32);      // DMA next chunk behind the math
    stageA(k0);                               // LN'd bf16 A tile (overlaps DMA)
    if (!last) asm volatile("s_wait_asynccnt 0x10" ::: "memory");  // current chunk landed
    else       asm volatile("s_wait_asynccnt 0x0"  ::: "memory");
    __syncthreads();
    FragU a = loadA();
    const unsigned char* brow = curPtr + (lane & 15) * 2 + nbase * 2;  // lane%16 = N column
    auto ldFrag = [&](int ni) {
      FragU f;
      #pragma unroll
      for (int s = 0; s < 16; ++s) {
        const int k = (s < 8) ? (koff + s) : (16 + koff + (s - 8));
        f.v[s] = *(const __bf16*)(brow + k * BROW + ni * 32);
      }
      return f;
    };
    // rolling 2-fragment window: ds_loads for ni+2 issue before the WMMA on ni,
    // so the emitted waits are partial (dscnt<=2) instead of full drains
    FragU fa = ldFrag(0);
    FragU fb = ldFrag(1);
    #pragma unroll
    for (int ni = 0; ni < 8; ++ni) {
      FragU fn = fb;
      if (ni + 2 < 8) fn = ldFrag(ni + 2);
      acc[ni] = __builtin_amdgcn_wmma_f32_16x16x32_bf16(
          false, a.v, false, fa.v, (short)0, acc[ni], false, false);
      fa = fb; fb = fn;
    }
    __syncthreads();                          // protect aTile + buffer reuse
  };

  for (int c = 0; c < 64; c += 2) {
    chunk(&bBuf[0][0], b1base, c * 32, false);
    chunk(&bBuf[1][0], b0base, (c + 1) * 32, c + 2 >= 64);
  }

  // epilogue: bias + exact GELU, D-layout -> row-major bf16 LDS tile
  const int mhi = (lane >> 4) * 8;
  #pragma unroll
  for (int ni = 0; ni < 8; ++ni) {
    const int n = nbase + ni * 16 + (lane & 15);
    #pragma unroll
    for (int v = 0; v < 8; ++v) {
      const float h = acc[ni][v] + bias[ni];
      const float g = 0.5f * h * (1.0f + erff(h * 0.70710678118654752f));
      hidbuf[(v + mhi) * 1024 + n] = f2bf(g);
    }
  }
  __syncthreads();

  // logits: one (token, head) per thread, both operands streamed as b128
  {
    const int m = tid >> 4, h = tid & 15;
    float s = r_b2[h];
    const uint4* hr = (const uint4*)&hidbuf[m * 1024];
    const uint4* wr = (const uint4*)(w2t + (size_t)h * 1024);
    for (int c = 0; c < 128; ++c) {
      const uint4 hq = hr[c], wq = wr[c];
      const unsigned hd[4] = { hq.x, hq.y, hq.z, hq.w };
      const unsigned wd[4] = { wq.x, wq.y, wq.z, wq.w };
      #pragma unroll
      for (int d = 0; d < 4; ++d) {
        s = fmaf(__uint_as_float(hd[d] << 16),         __uint_as_float(wd[d] << 16),         s);
        s = fmaf(__uint_as_float(hd[d] & 0xffff0000u), __uint_as_float(wd[d] & 0xffff0000u), s);
      }
    }
    logits_s[m * 16 + h] = s;
  }
  __syncthreads();

  // softmax + top-4 renorm + entropy, one thread per token
  if (tid < 16) {
    const int m = tid;
    float l[16], mx = -1e30f;
    #pragma unroll
    for (int h = 0; h < 16; ++h) { l[h] = logits_s[m * 16 + h]; mx = fmaxf(mx, l[h]); }
    float e[16], sum = 0.f;
    #pragma unroll
    for (int h = 0; h < 16; ++h) { e[h] = expf(l[h] - mx); sum += e[h]; }
    const float inv = 1.0f / sum;
    float a[16], tmp[16], keep[16];
    #pragma unroll
    for (int h = 0; h < 16; ++h) { a[h] = e[h] * inv; tmp[h] = a[h]; keep[h] = 0.f; }
    for (int r = 0; r < TOPK; ++r) {
      int bi = 0; float bv = tmp[0];
      #pragma unroll
      for (int h = 1; h < 16; ++h) if (tmp[h] > bv) { bv = tmp[h]; bi = h; }
      keep[bi] = 1.f; tmp[bi] = -1e30f;
    }
    float ms = 0.f;
    #pragma unroll
    for (int h = 0; h < 16; ++h) ms += a[h] * keep[h];
    const float inv2 = 1.0f / fmaxf(ms, 1e-12f);
    float ent = 0.f;
    float* ao = out + ((size_t)(b * SS + s0 + m)) * N_HEADS;
    #pragma unroll
    for (int h = 0; h < 16; ++h) {
      const float al = a[h] * keep[h] * inv2;
      ao[h] = al;
      ent += al * logf(fmaxf(al, 1e-12f));
    }
    atomicAdd(&out[ENT_OFF], -ent * (1.0f / (BB * SS)));
  }
}

// ---------------------------------------------------------------- host
extern "C" void kernel_launch(void* const* d_in, const int* in_sizes, int n_in,
                              void* d_out, int out_size, void* d_ws, size_t ws_size,
                              hipStream_t stream) {
  const float* x      = (const float*)d_in[0];
  const float* state  = (const float*)d_in[1];
  const float* pool_g = (const float*)d_in[2];
  const float* pool_b = (const float*)d_in[3];
  const float* inp_w  = (const float*)d_in[4];
  const float* inp_b  = (const float*)d_in[5];
  const float* wz_w   = (const float*)d_in[6];
  const float* wz_b   = (const float*)d_in[7];
  const float* wh_w   = (const float*)d_in[8];
  const float* wh_b   = (const float*)d_in[9];
  const float* lnh_g  = (const float*)d_in[10];
  const float* lnh_b  = (const float*)d_in[11];
  const float* rln_g  = (const float*)d_in[12];
  const float* rln_b  = (const float*)d_in[13];
  const float* r_w1   = (const float*)d_in[14];
  const float* r_b1   = (const float*)d_in[15];
  const float* r_w2   = (const float*)d_in[16];
  const float* r_b2   = (const float*)d_in[17];
  float* out = (float*)d_out;

  float* g_sum = (float*)d_ws;                                  // 8192 f32
  float* h_ws  = g_sum + BB * D_MODEL;                          // 8192 f32
  unsigned short* w1bf = (unsigned short*)(h_ws + BB * D_CTRL); // 2048*1024 bf16
  unsigned short* w2t  = w1bf + (size_t)D_CAT * D_CTRL;         // 16*1024 bf16 (transposed)

  init_kernel<<<(BB * D_MODEL + 255) / 256, 256, 0, stream>>>(g_sum, out);
  w1cvt_kernel<<<(D_CAT * D_CTRL) / (256 * 8), 256, 0, stream>>>(r_w1, w1bf);
  w2cvt_kernel<<<(N_HEADS * D_CTRL) / 256, 256, 0, stream>>>(r_w2, w2t);
  pool_kernel<<<dim3(SS / 64, BB), 256, 0, stream>>>(x, pool_g, pool_b, g_sum);
  ctrl_kernel<<<BB, 256, 0, stream>>>(g_sum, state, inp_w, inp_b, wz_w, wz_b,
                                      wh_w, wh_b, lnh_g, lnh_b, out, h_ws);
  router_kernel<<<dim3(SS / 16, BB), 256, 0, stream>>>(x, h_ws, rln_g, rln_b,
                                                       w1bf, r_b1, w2t, r_b2, out);
}
